// EncoderDecoder_43722767073459
// MI455X (gfx1250) — compile-verified, matrix operand order checked by reading, not run on hardware
//
#include <hip/hip_runtime.h>

typedef __bf16 bf16_t;
typedef __attribute__((ext_vector_type(16))) __bf16 v16bf;
typedef __attribute__((ext_vector_type(8)))  float  v8f;

#define HID   512
#define G4H   2048
#define ROWS  16
#define HPAD  8
#define HSTR  (HID + HPAD)
#define T_IN  48
#define TLEN  24
#define NSEQ  1024
#define WGT   256

union FragU { v16bf v; uint4 q[2]; };

__device__ __forceinline__ v16bf load_frag(const bf16_t* p) {
  FragU f;
  f.q[0] = *(const uint4*)(p);
  f.q[1] = *(const uint4*)(p + 16);
  return f.v;
}

__device__ __forceinline__ float sigf(float x) {
  return 1.0f / (1.0f + __expf(-x));
}
__device__ __forceinline__ float tanh_fast(float x) {
  float e = __expf(-2.0f * fabsf(x));
  float t = (1.0f - e) / (1.0f + e);
  return copysignf(t, x);
}

// Accumulate one 16x16 D tile per gate for hidden-col tile `ht`:
//   acc[g] += A(h tile, 16x512) x B(W rows g*512+ht*16 .. +15, K=512)
__device__ __forceinline__ void gemm_ht(const bf16_t* __restrict__ hbuf,  // LDS [16][HSTR] bf16
                                        const bf16_t* __restrict__ W,    // global [2048][512] bf16
                                        int ht, int lo, int hi, v8f acc[4]) {
  const bf16_t* arow = hbuf + lo * HSTR + 8 * hi;
#pragma unroll 4
  for (int kb = 0; kb < HID / 32; ++kb) {
    v16bf a = load_frag(arow + kb * 32);
#pragma unroll
    for (int g = 0; g < 4; ++g) {
      const bf16_t* brow = W + (size_t)(g * HID + ht * 16 + lo) * HID + kb * 32 + 8 * hi;
      v16bf b = load_frag(brow);
      acc[g] = __builtin_amdgcn_wmma_f32_16x16x32_bf16(false, a, false, b,
                                                       (short)0, acc[g], false, false);
    }
  }
}

// layer with scalar input (enc layer0: x, dec layer0: fed-back y):
// z = xv[m]*wihcol[n] + h@Whh^T + bias ; update c (regs) and h (LDS, bf16)
__device__ __forceinline__ void layer0_step(bf16_t* hbuf, const bf16_t* __restrict__ Whh,
                                            const float* __restrict__ wihcol,
                                            const float* __restrict__ bias,
                                            const float* xv, v8f* cst,
                                            int wave, int lo, int hi) {
  float hv[4][8];
#pragma unroll
  for (int j = 0; j < 4; ++j) {
    const int ht = wave + 8 * j;
    v8f acc[4];
#pragma unroll
    for (int g = 0; g < 4; ++g) acc[g] = (v8f){0.f,0.f,0.f,0.f,0.f,0.f,0.f,0.f};
    gemm_ht(hbuf, Whh, ht, lo, hi, acc);
#pragma unroll
    for (int g = 0; g < 4; ++g) {
      const int n = g * HID + ht * 16 + lo;
      const float bb = bias[n];
      const float wx = wihcol[n];
#pragma unroll
      for (int r = 0; r < 8; ++r)
        acc[g][r] = acc[g][r] + bb + xv[r + 8 * hi] * wx;
    }
#pragma unroll
    for (int r = 0; r < 8; ++r) {
      float gi = sigf(acc[0][r]);
      float gf = sigf(acc[1][r]);
      float gg = tanh_fast(acc[2][r]);
      float go = sigf(acc[3][r]);
      float cn = gf * cst[j][r] + gi * gg;
      cst[j][r] = cn;
      hv[j][r] = go * tanh_fast(cn);
    }
  }
  __syncthreads();   // all waves done reading old h
#pragma unroll
  for (int j = 0; j < 4; ++j) {
    const int ht = wave + 8 * j;
#pragma unroll
    for (int r = 0; r < 8; ++r)
      hbuf[(r + 8 * hi) * HSTR + ht * 16 + lo] = (bf16_t)hv[j][r];
  }
  __syncthreads();   // new h visible
}

// layer1: z = h0@Wih^T + h1@Whh^T + bias
__device__ __forceinline__ void layer1_step(const bf16_t* h0buf, bf16_t* h1buf,
                                            const bf16_t* __restrict__ Wih,
                                            const bf16_t* __restrict__ Whh,
                                            const float* __restrict__ bias,
                                            v8f* cst, int wave, int lo, int hi) {
  float hv[4][8];
#pragma unroll
  for (int j = 0; j < 4; ++j) {
    const int ht = wave + 8 * j;
    v8f acc[4];
#pragma unroll
    for (int g = 0; g < 4; ++g) acc[g] = (v8f){0.f,0.f,0.f,0.f,0.f,0.f,0.f,0.f};
    gemm_ht(h0buf, Wih, ht, lo, hi, acc);
    gemm_ht(h1buf, Whh, ht, lo, hi, acc);
#pragma unroll
    for (int g = 0; g < 4; ++g) {
      const float bb = bias[g * HID + ht * 16 + lo];
#pragma unroll
      for (int r = 0; r < 8; ++r) acc[g][r] += bb;
    }
#pragma unroll
    for (int r = 0; r < 8; ++r) {
      float gi = sigf(acc[0][r]);
      float gf = sigf(acc[1][r]);
      float gg = tanh_fast(acc[2][r]);
      float go = sigf(acc[3][r]);
      float cn = gf * cst[j][r] + gi * gg;
      cst[j][r] = cn;
      hv[j][r] = go * tanh_fast(cn);
    }
  }
  __syncthreads();
#pragma unroll
  for (int j = 0; j < 4; ++j) {
    const int ht = wave + 8 * j;
#pragma unroll
    for (int r = 0; r < 8; ++r)
      h1buf[(r + 8 * hi) * HSTR + ht * 16 + lo] = (bf16_t)hv[j][r];
  }
  __syncthreads();
}

__global__ __launch_bounds__(WGT, 1)
void lstm_seq2seq_kernel(const float* __restrict__ x,       // [1024,48]
                         const float* __restrict__ wih0e,   // [2048]
                         const float* __restrict__ wih0d,   // [2048]
                         const float* __restrict__ fcW,     // [512]
                         const float* __restrict__ fcB,     // [1]
                         const bf16_t* __restrict__ whh0e,
                         const bf16_t* __restrict__ wih1e,
                         const bf16_t* __restrict__ whh1e,
                         const bf16_t* __restrict__ whh0d,
                         const bf16_t* __restrict__ wih1d,
                         const bf16_t* __restrict__ whh1d,
                         const float* __restrict__ be0,
                         const float* __restrict__ be1,
                         const float* __restrict__ bd0,
                         const float* __restrict__ bd1,
                         float* __restrict__ out)            // [1024,24]
{
  __shared__ __align__(16) bf16_t h0s[ROWS * HSTR];
  __shared__ __align__(16) bf16_t h1s[ROWS * HSTR];
  __shared__ float xs[ROWS];
  __shared__ float red[ROWS * 16];

  const int tid  = threadIdx.x;
  const int wave = tid >> 5;
  const int lane = tid & 31;
  const int lo   = lane & 15;
  const int hi   = lane >> 4;
  const int r0   = blockIdx.x * ROWS;

  for (int i = tid; i < ROWS * HSTR; i += WGT) {
    h0s[i] = (bf16_t)0.0f;
    h1s[i] = (bf16_t)0.0f;
  }
  v8f c0st[4], c1st[4];
#pragma unroll
  for (int j = 0; j < 4; ++j) {
    c0st[j] = (v8f){0.f,0.f,0.f,0.f,0.f,0.f,0.f,0.f};
    c1st[j] = (v8f){0.f,0.f,0.f,0.f,0.f,0.f,0.f,0.f};
  }
  __syncthreads();

  // ---------------- encoder: 48 steps, 2 layers interleaved ----------------
  for (int t = 0; t < T_IN; ++t) {
    if (tid < ROWS) xs[tid] = x[(size_t)(r0 + tid) * T_IN + t];
    __syncthreads();
    layer0_step(h0s, whh0e, wih0e, be0, xs, c0st, wave, lo, hi);
    layer1_step(h0s, h1s, wih1e, whh1e, be1, c1st, wave, lo, hi);
  }

  // ---------------- decoder: 24 steps, states carry over -------------------
  if (tid < ROWS) xs[tid] = 0.0f;   // y0 = 0
  __syncthreads();
  const float fcb = fcB[0];
  for (int t = 0; t < TLEN; ++t) {
    layer0_step(h0s, whh0d, wih0d, bd0, xs, c0st, wave, lo, hi);
    layer1_step(h0s, h1s, wih1d, whh1d, bd1, c1st, wave, lo, hi);
    // fc: y[m] = h1[m,:] . fcW + fcb  (256 threads -> 16x16 partials)
    {
      const int m = tid >> 4, seg = tid & 15;
      float p = 0.0f;
#pragma unroll 8
      for (int q = 0; q < 32; ++q) {
        const int col = seg * 32 + q;
        p += (float)h1s[m * HSTR + col] * fcW[col];
      }
      red[m * 16 + seg] = p;
    }
    __syncthreads();
    if (tid < ROWS) {
      float s = fcb;
#pragma unroll
      for (int q = 0; q < 16; ++q) s += red[tid * 16 + q];
      xs[tid] = s;                                   // feedback y
      out[(size_t)(r0 + tid) * TLEN + t] = s;
    }
    __syncthreads();
  }
}

// ---- pre-pass: f32 weights -> bf16 (L2-resident), fuse bias pairs --------
__global__ void cvt_bf16_kernel(const float* __restrict__ src, bf16_t* __restrict__ dst, int n) {
  int i = blockIdx.x * 256 + threadIdx.x;
  if (i < n) dst[i] = (bf16_t)src[i];
}
__global__ void bias_sum_kernel(const float* __restrict__ a, const float* __restrict__ b,
                                float* __restrict__ o, int n) {
  int i = blockIdx.x * 256 + threadIdx.x;
  if (i < n) o[i] = a[i] + b[i];
}

extern "C" void kernel_launch(void* const* d_in, const int* in_sizes, int n_in,
                              void* d_out, int out_size, void* d_ws, size_t ws_size,
                              hipStream_t stream) {
  const float* x     = (const float*)d_in[0];
  const float* eWih0 = (const float*)d_in[1];
  const float* eWhh0 = (const float*)d_in[2];
  const float* eBih0 = (const float*)d_in[3];
  const float* eBhh0 = (const float*)d_in[4];
  const float* eWih1 = (const float*)d_in[5];
  const float* eWhh1 = (const float*)d_in[6];
  const float* eBih1 = (const float*)d_in[7];
  const float* eBhh1 = (const float*)d_in[8];
  const float* dWih0 = (const float*)d_in[9];
  const float* dWhh0 = (const float*)d_in[10];
  const float* dBih0 = (const float*)d_in[11];
  const float* dBhh0 = (const float*)d_in[12];
  const float* dWih1 = (const float*)d_in[13];
  const float* dWhh1 = (const float*)d_in[14];
  const float* dBih1 = (const float*)d_in[15];
  const float* dBhh1 = (const float*)d_in[16];
  const float* fcW   = (const float*)d_in[17];
  const float* fcB   = (const float*)d_in[18];

  const int WN = G4H * HID;  // 1,048,576 elems per big matrix
  char* ws = (char*)d_ws;
  bf16_t* whh0e = (bf16_t*)ws; ws += (size_t)WN * sizeof(bf16_t);
  bf16_t* wih1e = (bf16_t*)ws; ws += (size_t)WN * sizeof(bf16_t);
  bf16_t* whh1e = (bf16_t*)ws; ws += (size_t)WN * sizeof(bf16_t);
  bf16_t* whh0d = (bf16_t*)ws; ws += (size_t)WN * sizeof(bf16_t);
  bf16_t* wih1d = (bf16_t*)ws; ws += (size_t)WN * sizeof(bf16_t);
  bf16_t* whh1d = (bf16_t*)ws; ws += (size_t)WN * sizeof(bf16_t);
  float* be0 = (float*)ws; ws += G4H * sizeof(float);
  float* be1 = (float*)ws; ws += G4H * sizeof(float);
  float* bd0 = (float*)ws; ws += G4H * sizeof(float);
  float* bd1 = (float*)ws; ws += G4H * sizeof(float);

  const dim3 blk(256);
  const dim3 gw((WN + 255) / 256);
  cvt_bf16_kernel<<<gw, blk, 0, stream>>>(eWhh0, whh0e, WN);
  cvt_bf16_kernel<<<gw, blk, 0, stream>>>(eWih1, wih1e, WN);
  cvt_bf16_kernel<<<gw, blk, 0, stream>>>(eWhh1, whh1e, WN);
  cvt_bf16_kernel<<<gw, blk, 0, stream>>>(dWhh0, whh0d, WN);
  cvt_bf16_kernel<<<gw, blk, 0, stream>>>(dWih1, wih1d, WN);
  cvt_bf16_kernel<<<gw, blk, 0, stream>>>(dWhh1, whh1d, WN);
  const dim3 gb((G4H + 255) / 256);
  bias_sum_kernel<<<gb, blk, 0, stream>>>(eBih0, eBhh0, be0, G4H);
  bias_sum_kernel<<<gb, blk, 0, stream>>>(eBih1, eBhh1, be1, G4H);
  bias_sum_kernel<<<gb, blk, 0, stream>>>(dBih0, dBhh0, bd0, G4H);
  bias_sum_kernel<<<gb, blk, 0, stream>>>(dBih1, dBhh1, bd1, G4H);

  lstm_seq2seq_kernel<<<NSEQ / ROWS, WGT, 0, stream>>>(
      x, eWih0, dWih0, fcW, fcB,
      whh0e, wih1e, whh1e, whh0d, wih1d, whh1d,
      be0, be1, bd0, bd1, (float*)d_out);
}
